// ReconstructionLSTM_85813446574179
// MI455X (gfx1250) — matrix operand order repaired and checked by reading, not compiled, not run on hardware
//
#include <hip/hip_runtime.h>
#include <hip/hip_bf16.h>
#include <cstdint>
#include <cstddef>

// ---------------------------------------------------------------------------
// MI455X (gfx1250) implementation of ReconstructionLSTM.
//   * encoder recurrence + input projection fused, bf16 WMMA / f32 accum
//   * kproj GEMM via bf16 WMMA
//   * decoder attention/cell as per-step f32 VALU kernels (graph-replayed)
//   * sigmoid/tanh/exp via CDNA5 hardware transcendentals
//     (v_tanh_f32 / v_exp_f32 / v_rcp_f32) instead of branchy libm
// ---------------------------------------------------------------------------

typedef __bf16 bf16_t;
typedef __attribute__((ext_vector_type(16))) __bf16 v16bf;
typedef __attribute__((ext_vector_type(8)))  float  v8f;

#define H_    64
#define NL_   5
#define DF_   24
#define EMB_  8
#define B_    256
#define S_    256
#define NSEQ_ (B_ * NL_)   // 1280
#define XEW_  32           // DF + EMB
#define G4_   256          // 4*H

#define LOG2E_ 1.44269504088896340736f

// --- CDNA5 hardware transcendentals ----------------------------------------
__device__ __forceinline__ float fast_exp(float x) {
  return __builtin_amdgcn_exp2f(x * LOG2E_);          // v_exp_f32
}
__device__ __forceinline__ float fast_sig(float x) {
  // v_exp_f32 saturates cleanly (0 / +inf), so no range guards needed.
  return __builtin_amdgcn_rcpf(1.0f + __builtin_amdgcn_exp2f(-LOG2E_ * x));
}
__device__ __forceinline__ float fast_tanh(float x) {
#if __has_builtin(__builtin_amdgcn_tanhf)
  return __builtin_amdgcn_tanhf(x);                   // v_tanh_f32
#else
  float r;
  asm("v_tanh_f32 %0, %1" : "=v"(r) : "v"(x));
  return r;
#endif
}

__device__ __forceinline__ v8f wmma_bf16(v16bf a, v16bf b, v8f c) {
  // (neg_a, A, neg_b, B, c_mod, C, reuse_a, reuse_b) -> v_wmma_f32_16x16x32_bf16
  return __builtin_amdgcn_wmma_f32_16x16x32_bf16(false, a, false, b, (short)0, c, false, false);
}

// B fragment (K x 16, 16-bit): lanes 0-15 hold K=kb..kb+15 of col (lane&15);
// lanes 16-31 hold K=kb+16..kb+31.  W is row-major (out_col, K).
__device__ __forceinline__ v16bf bfragB(const float* __restrict__ W, int Kdim,
                                        int col, int kbase) {
  v16bf r;
#pragma unroll
  for (int e = 0; e < 16; ++e) r[e] = (bf16_t)W[col * Kdim + kbase + e];
  return r;
}

// A fragment (16 x K, 16-bit): per-lane elements 0..15 -> K=kbase..kbase+15
// (kbase already includes the lane-half +16).
__device__ __forceinline__ v16bf fragA(const bf16_t* __restrict__ row, int kbase) {
  v16bf r;
#pragma unroll
  for (int e = 0; e < 16; ++e) r[e] = row[kbase + e];
  return r;
}

// ---------------------------------------------------------------------------
// K0: lang_active + x_flat (bf16).  One block per (b,l) sequence.
// ---------------------------------------------------------------------------
__global__ __launch_bounds__(256) void prep_kernel(const float* __restrict__ x,
                                                   const float* __restrict__ lemb,
                                                   bf16_t* __restrict__ xe) {
  const int b = blockIdx.x / NL_;
  const int l = blockIdx.x % NL_;
  __shared__ float red[256];
  float s = 0.0f;
  for (int i = threadIdx.x; i < S_ * DF_; i += 256) {
    int si = i / DF_, d = i % DF_;
    s += fabsf(x[(((size_t)b * S_ + si) * NL_ + l) * DF_ + d]);
  }
  red[threadIdx.x] = s;
  __syncthreads();
  for (int off = 128; off > 0; off >>= 1) {
    if (threadIdx.x < off) red[threadIdx.x] += red[threadIdx.x + off];
    __syncthreads();
  }
  const float active = (red[0] > 0.0f) ? 1.0f : 0.0f;
  const int si = threadIdx.x;  // one s per thread
  bf16_t* dst = xe + ((size_t)blockIdx.x * S_ + si) * XEW_;
  const float* src = x + (((size_t)b * S_ + si) * NL_ + l) * DF_;
#pragma unroll
  for (int d = 0; d < DF_; ++d) dst[d] = (bf16_t)src[d];
#pragma unroll
  for (int j = 0; j < EMB_; ++j) dst[DF_ + j] = (bf16_t)(lemb[l * EMB_ + j] * active);
}

// ---------------------------------------------------------------------------
// K1: bidirectional encoder LSTM.  grid = (NSEQ/16, 2 directions).
// Block: 16 batch rows, 8 waves; wave w owns gate columns [32w, 32w+32).
// Weight B-fragments live in VGPRs across all 256 steps; next-step x fragment
// is software-pipelined past the cell-update phase.
// ---------------------------------------------------------------------------
__global__ __launch_bounds__(256) void enc_kernel(
    const bf16_t* __restrict__ xe,
    const float* __restrict__ Wih_f, const float* __restrict__ Whh_f, const float* __restrict__ b_f,
    const float* __restrict__ Wih_r, const float* __restrict__ Whh_r, const float* __restrict__ b_r,
    float* __restrict__ out_f, float* __restrict__ out_r,
    float* __restrict__ hf, float* __restrict__ cf,
    float* __restrict__ hr, float* __restrict__ cr) {
  const int dir = blockIdx.y;  // 0 fwd, 1 rev
  const int n0 = blockIdx.x * 16;
  const float* Wih = dir ? Wih_r : Wih_f;
  const float* Whh = dir ? Whh_r : Whh_f;
  const float* bias = dir ? b_r : b_f;
  float* outbuf = dir ? out_r : out_f;
  float* hT = dir ? hr : hf;
  float* cT = dir ? cr : cf;

  __shared__ bf16_t h_bf[16][H_];
  __shared__ float h_fs[16][H_];
  __shared__ float c_fs[16][H_];
  __shared__ float gates[16][G4_];

  const int tid = threadIdx.x;
  const int wave = tid >> 5;
  const int lane = tid & 31;
  const int g0 = wave * 32;

  for (int i = tid; i < 16 * H_; i += 256) {
    (&h_bf[0][0])[i] = (bf16_t)0.0f;
    (&h_fs[0][0])[i] = 0.0f;
    (&c_fs[0][0])[i] = 0.0f;
  }

  const int colA = g0 + (lane & 15);
  const int colB = g0 + 16 + (lane & 15);
  const int kB = (lane & 16) ? 16 : 0;   // B-fragment K base (within 32-block)
  const int kA = (lane & 16) ? 16 : 0;   // A-fragment K base (within 32-block)

  // Persistent weight fragments (f32 -> bf16 once).
  v16bf Bih0 = bfragB(Wih, XEW_, colA, kB);
  v16bf Bih1 = bfragB(Wih, XEW_, colB, kB);
  v16bf Bhh0a = bfragB(Whh, H_, colA, kB);
  v16bf Bhh0b = bfragB(Whh, H_, colA, 32 + kB);
  v16bf Bhh1a = bfragB(Whh, H_, colB, kB);
  v16bf Bhh1b = bfragB(Whh, H_, colB, 32 + kB);
  const float bias0 = bias[colA];
  const float bias1 = bias[colB];

  const int arow = lane & 15;
  const bf16_t* xrow = xe + ((size_t)(n0 + arow)) * S_ * XEW_;

  __syncthreads();

  // software pipeline: preload x fragment for the first step
  v16bf Ax = fragA(xrow + (size_t)(dir ? (S_ - 1) : 0) * XEW_, kA);

  for (int t = 0; t < S_; ++t) {
    const int s = dir ? (S_ - 1 - t) : t;
    if (t + 1 < S_) {
      const int sn = dir ? (S_ - 2 - t) : (t + 1);
      __builtin_prefetch(xrow + (size_t)sn * XEW_, 0, 1);  // global_prefetch_b8
    }
    v16bf Ah0 = fragA(&h_bf[arow][0], kA);
    v16bf Ah1 = fragA(&h_bf[arow][0], 32 + kA);

    v8f acc0, acc1;
#pragma unroll
    for (int j = 0; j < 8; ++j) { acc0[j] = bias0; acc1[j] = bias1; }
    acc0 = wmma_bf16(Ax, Bih0, acc0);
    acc0 = wmma_bf16(Ah0, Bhh0a, acc0);
    acc0 = wmma_bf16(Ah1, Bhh0b, acc0);
    acc1 = wmma_bf16(Ax, Bih1, acc1);
    acc1 = wmma_bf16(Ah0, Bhh1a, acc1);
    acc1 = wmma_bf16(Ah1, Bhh1b, acc1);

    // issue next-step x load so its latency overlaps the cell update
    if (t + 1 < S_) {
      const int sn = dir ? (S_ - 2 - t) : (t + 1);
      Ax = fragA(xrow + (size_t)sn * XEW_, kA);
    }

    const int mofs = (lane & 16) ? 8 : 0;
    const int ncol = lane & 15;
#pragma unroll
    for (int j = 0; j < 8; ++j) {
      gates[mofs + j][g0 + ncol] = acc0[j];
      gates[mofs + j][g0 + 16 + ncol] = acc1[j];
    }
    __syncthreads();

    // cell update: 16 rows x 64 hidden, 4 per thread, all f32
#pragma unroll
    for (int r = 0; r < 4; ++r) {
      const int idx = tid + r * 256;
      const int row = idx >> 6, hh = idx & 63;
      const float gi = gates[row][hh];
      const float gf = gates[row][H_ + hh];
      const float gg = gates[row][2 * H_ + hh];
      const float go = gates[row][3 * H_ + hh];
      const float cc = fast_sig(gf) * c_fs[row][hh] + fast_sig(gi) * fast_tanh(gg);
      const float hv = fast_sig(go) * fast_tanh(cc);
      c_fs[row][hh] = cc;
      h_fs[row][hh] = hv;
      h_bf[row][hh] = (bf16_t)hv;
      outbuf[((size_t)(n0 + row)) * S_ * H_ + (size_t)s * H_ + hh] = hv;
    }
    __syncthreads();
  }

  for (int i = tid; i < 16 * H_; i += 256) {
    const int row = i >> 6, hh = i & 63;
    hT[(n0 + row) * H_ + hh] = h_fs[row][hh];
    cT[(n0 + row) * H_ + hh] = c_fs[row][hh];
  }
}

// ---------------------------------------------------------------------------
// K2: enc_out = mean over l of [out_f | out_r]  (f32 + bf16 copy)
// ---------------------------------------------------------------------------
__global__ __launch_bounds__(256) void reduce_kernel(const float* __restrict__ out_f,
                                                     const float* __restrict__ out_r,
                                                     float* __restrict__ enc,
                                                     bf16_t* __restrict__ enc_bf) {
  const size_t idx = (size_t)blockIdx.x * 256 + threadIdx.x;  // B*S*128
  const int d = (int)(idx & 127);
  const size_t bs = idx >> 7;
  const int s = (int)(bs & 255);
  const int b = (int)(bs >> 8);
  const float* src = (d < H_) ? out_f : out_r;
  const int dd = d & 63;
  float sum = 0.0f;
#pragma unroll
  for (int l = 0; l < NL_; ++l)
    sum += src[((size_t)(b * NL_ + l) * S_ + s) * H_ + dd];
  const float m = sum * 0.2f;
  enc[idx] = m;
  enc_bf[idx] = (bf16_t)m;
}

// ---------------------------------------------------------------------------
// K3: kproj = enc_out @ Wk^T + bk   (M=65536, K=128, N=128) via bf16 WMMA
// ---------------------------------------------------------------------------
__global__ __launch_bounds__(256) void kproj_kernel(const bf16_t* __restrict__ enc_bf,
                                                    const float* __restrict__ Wk,
                                                    const float* __restrict__ bk,
                                                    float* __restrict__ kproj) {
  const int m0 = blockIdx.x * 16;
  const int wave = threadIdx.x >> 5, lane = threadIdx.x & 31;
  const int g0 = wave * 16;  // 8 waves x 16 = 128 out cols
  const int kB = (lane & 16) ? 16 : 0;
  const int kA = (lane & 16) ? 16 : 0;
  const int col = g0 + (lane & 15);

  v16bf B0 = bfragB(Wk, 2 * H_, col, kB);
  v16bf B1 = bfragB(Wk, 2 * H_, col, 32 + kB);
  v16bf B2 = bfragB(Wk, 2 * H_, col, 64 + kB);
  v16bf B3 = bfragB(Wk, 2 * H_, col, 96 + kB);
  const float bb = bk[col];

  const bf16_t* arow = enc_bf + ((size_t)(m0 + (lane & 15))) * (2 * H_);
  v16bf A0 = fragA(arow, kA);
  v16bf A1 = fragA(arow, 32 + kA);
  v16bf A2 = fragA(arow, 64 + kA);
  v16bf A3 = fragA(arow, 96 + kA);

  v8f acc;
#pragma unroll
  for (int j = 0; j < 8; ++j) acc[j] = bb;
  acc = wmma_bf16(A0, B0, acc);
  acc = wmma_bf16(A1, B1, acc);
  acc = wmma_bf16(A2, B2, acc);
  acc = wmma_bf16(A3, B3, acc);

  const int mofs = (lane & 16) ? 8 : 0;
  const int ncol = lane & 15;
#pragma unroll
  for (int j = 0; j < 8; ++j)
    kproj[((size_t)(m0 + mofs + j)) * (2 * H_) + g0 + ncol] = acc[j];
}

// ---------------------------------------------------------------------------
// K4: decoder initial state  h_d/c_d = [hf_m|hr_m]@hb_W^T+b ; curr = 0
// Block: 4 b's x 64 hidden.
// ---------------------------------------------------------------------------
__global__ __launch_bounds__(256) void hd_kernel(
    const float* __restrict__ hf, const float* __restrict__ hr,
    const float* __restrict__ cf, const float* __restrict__ cr,
    const float* __restrict__ hbW, const float* __restrict__ hbB,
    const float* __restrict__ cbW, const float* __restrict__ cbB,
    float* __restrict__ h_dec, float* __restrict__ c_dec, float* __restrict__ curr) {
  const int b0 = blockIdx.x * 4;
  __shared__ float hm[4][2 * H_];
  __shared__ float cm[4][2 * H_];
  for (int i = threadIdx.x; i < 4 * 2 * H_; i += 256) {
    const int bb = i >> 7, d = i & 127;
    const int dd = d & 63;
    float sh = 0.0f, sc = 0.0f;
#pragma unroll
    for (int l = 0; l < NL_; ++l) {
      const size_t n = (size_t)(b0 + bb) * NL_ + l;
      sh += (d < H_) ? hf[n * H_ + dd] : hr[n * H_ + dd];
      sc += (d < H_) ? cf[n * H_ + dd] : cr[n * H_ + dd];
    }
    hm[bb][d] = sh * 0.2f;
    cm[bb][d] = sc * 0.2f;
  }
  __syncthreads();
  const int bb = threadIdx.x >> 6, j = threadIdx.x & 63;
  float ah = hbB[j], ac = cbB[j];
  for (int k = 0; k < 2 * H_; ++k) {
    ah += hbW[j * 2 * H_ + k] * hm[bb][k];
    ac += cbW[j * 2 * H_ + k] * cm[bb][k];
  }
  h_dec[(b0 + bb) * H_ + j] = ah;
  c_dec[(b0 + bb) * H_ + j] = ac;
  if (j < DF_) curr[(b0 + bb) * DF_ + j] = 0.0f;
}

// ---------------------------------------------------------------------------
// K5 (per step): scores -> exp.  Block = one b, thread = one s.
// tanh-bounded scores (|score| <~ 6) -> exp without max subtraction is safe.
// Inner loop = 128 x (v_tanh_f32 + fma): pure CDNA5 trans throughput.
// ---------------------------------------------------------------------------
__global__ __launch_bounds__(256) void score_kernel(
    const float* __restrict__ h_dec, const float* __restrict__ WqW,
    const float* __restrict__ Wqb, const float* __restrict__ kproj,
    const float* __restrict__ vW, const float* __restrict__ vb,
    float* __restrict__ ebuf) {
  const int b = blockIdx.x;
  __shared__ float q[2 * H_];
  __shared__ float hsh[H_];
  if (threadIdx.x < H_) hsh[threadIdx.x] = h_dec[b * H_ + threadIdx.x];
  __syncthreads();
  if (threadIdx.x < 2 * H_) {
    float a = Wqb[threadIdx.x];
    for (int k = 0; k < H_; ++k) a += WqW[threadIdx.x * H_ + k] * hsh[k];
    q[threadIdx.x] = a;
  }
  __syncthreads();
  const int s = threadIdx.x;
  const float* kp = kproj + ((size_t)b * S_ + s) * (2 * H_);
  float sc = vb[0];
  for (int d = 0; d < 2 * H_; ++d) sc += fast_tanh(q[d] + kp[d]) * vW[d];
  ebuf[b * S_ + s] = fast_exp(sc);
}

// ---------------------------------------------------------------------------
// K6 (per step): ctx[b,d] = sum_s e*enc / sum_s e.  Block = one b, 128 thr.
// ---------------------------------------------------------------------------
__global__ __launch_bounds__(128) void ctx_kernel(const float* __restrict__ ebuf,
                                                  const float* __restrict__ enc,
                                                  float* __restrict__ ctx) {
  const int b = blockIdx.x, d = threadIdx.x;
  const float* e = ebuf + b * S_;
  const float* en = enc + (size_t)b * S_ * (2 * H_) + d;
  float acc = 0.0f, den = 0.0f;
  for (int s = 0; s < S_; ++s) {
    const float w = e[s];
    den += w;
    acc += w * en[(size_t)s * (2 * H_)];
  }
  ctx[b * (2 * H_) + d] = acc * __builtin_amdgcn_rcpf(den);
}

// ---------------------------------------------------------------------------
// K7 (per step): decoder cell + fc head.  Block = one b, thread = one gate.
// ---------------------------------------------------------------------------
__global__ __launch_bounds__(256) void cell_kernel(
    const float* __restrict__ ctx, const float* __restrict__ dWih,
    const float* __restrict__ dWhh, const float* __restrict__ db,
    const float* __restrict__ fcW, const float* __restrict__ fcb,
    float* __restrict__ h_dec, float* __restrict__ c_dec,
    float* __restrict__ curr, float* __restrict__ out, int t) {
  const int b = blockIdx.x;
  const int tid = threadIdx.x;
  __shared__ float inv[DF_ + 2 * H_];  // [curr | ctx] = 152
  __shared__ float hsh[H_];
  __shared__ float gsh[G4_];
  __shared__ float hnew[H_];
  if (tid < DF_) inv[tid] = curr[b * DF_ + tid];
  else if (tid < DF_ + 2 * H_) inv[tid] = ctx[b * (2 * H_) + tid - DF_];
  else if (tid < DF_ + 2 * H_ + H_) hsh[tid - (DF_ + 2 * H_)] = h_dec[b * H_ + tid - (DF_ + 2 * H_)];
  __syncthreads();
  float g = db[tid];
  const float* wr = dWih + tid * (DF_ + 2 * H_);
  for (int k = 0; k < DF_ + 2 * H_; ++k) g += wr[k] * inv[k];
  const float* wh = dWhh + tid * H_;
  for (int k = 0; k < H_; ++k) g += wh[k] * hsh[k];
  gsh[tid] = g;
  __syncthreads();
  if (tid < H_) {
    const float gi = gsh[tid], gf = gsh[H_ + tid], gg = gsh[2 * H_ + tid], go = gsh[3 * H_ + tid];
    const float cc = fast_sig(gf) * c_dec[b * H_ + tid] + fast_sig(gi) * fast_tanh(gg);
    const float hv = fast_sig(go) * fast_tanh(cc);
    c_dec[b * H_ + tid] = cc;
    h_dec[b * H_ + tid] = hv;
    hnew[tid] = hv;
  }
  __syncthreads();
  if (tid < 25) {
    float r = fcb[tid];
    const float* fw = fcW + tid * H_;
    for (int k = 0; k < H_; ++k) r += fw[k] * hnew[k];
    const float val = (tid < DF_) ? fast_tanh(r) : fast_sig(r);
    out[((size_t)b * S_ + t) * 25 + tid] = val;
    if (tid < DF_) curr[b * DF_ + tid] = val;
  }
}

// ---------------------------------------------------------------------------
extern "C" void kernel_launch(void* const* d_in, const int* in_sizes, int n_in,
                              void* d_out, int out_size, void* d_ws, size_t ws_size,
                              hipStream_t stream) {
  (void)in_sizes; (void)n_in; (void)out_size; (void)ws_size;
  const float* x    = (const float*)d_in[0];
  const float* lemb = (const float*)d_in[1];
  const float* Wih_f = (const float*)d_in[2];
  const float* Whh_f = (const float*)d_in[3];
  const float* b_f   = (const float*)d_in[4];
  const float* Wih_r = (const float*)d_in[5];
  const float* Whh_r = (const float*)d_in[6];
  const float* b_r   = (const float*)d_in[7];
  const float* hbW   = (const float*)d_in[8];
  const float* hbB   = (const float*)d_in[9];
  const float* cbW   = (const float*)d_in[10];
  const float* cbB   = (const float*)d_in[11];
  const float* WqW   = (const float*)d_in[12];
  const float* Wqb   = (const float*)d_in[13];
  const float* WkW   = (const float*)d_in[14];
  const float* Wkb   = (const float*)d_in[15];
  const float* vW    = (const float*)d_in[16];
  const float* vb    = (const float*)d_in[17];
  const float* dWih  = (const float*)d_in[18];
  const float* dWhh  = (const float*)d_in[19];
  const float* db    = (const float*)d_in[20];
  const float* fcW   = (const float*)d_in[21];
  const float* fcb   = (const float*)d_in[22];
  float* out = (float*)d_out;

  // Workspace carve-up (~276 MB total)
  char* p = (char*)d_ws;
  auto take = [&](size_t bytes) -> void* {
    void* r = (void*)p;
    p += (bytes + 255) & ~(size_t)255;
    return r;
  };
  bf16_t* xe    = (bf16_t*)take((size_t)NSEQ_ * S_ * XEW_ * sizeof(bf16_t));
  float* out_f  = (float*)take((size_t)NSEQ_ * S_ * H_ * sizeof(float));
  float* out_r  = (float*)take((size_t)NSEQ_ * S_ * H_ * sizeof(float));
  float* hf     = (float*)take((size_t)NSEQ_ * H_ * sizeof(float));
  float* cf     = (float*)take((size_t)NSEQ_ * H_ * sizeof(float));
  float* hr     = (float*)take((size_t)NSEQ_ * H_ * sizeof(float));
  float* cr     = (float*)take((size_t)NSEQ_ * H_ * sizeof(float));
  float* enc    = (float*)take((size_t)B_ * S_ * 2 * H_ * sizeof(float));
  bf16_t* enc_bf = (bf16_t*)take((size_t)B_ * S_ * 2 * H_ * sizeof(bf16_t));
  float* kproj  = (float*)take((size_t)B_ * S_ * 2 * H_ * sizeof(float));
  float* h_dec  = (float*)take((size_t)B_ * H_ * sizeof(float));
  float* c_dec  = (float*)take((size_t)B_ * H_ * sizeof(float));
  float* curr   = (float*)take((size_t)B_ * DF_ * sizeof(float));
  float* ebuf   = (float*)take((size_t)B_ * S_ * sizeof(float));
  float* ctx    = (float*)take((size_t)B_ * 2 * H_ * sizeof(float));

  prep_kernel<<<NSEQ_, 256, 0, stream>>>(x, lemb, xe);
  enc_kernel<<<dim3(NSEQ_ / 16, 2), 256, 0, stream>>>(
      xe, Wih_f, Whh_f, b_f, Wih_r, Whh_r, b_r, out_f, out_r, hf, cf, hr, cr);
  reduce_kernel<<<(B_ * S_ * 2 * H_) / 256, 256, 0, stream>>>(out_f, out_r, enc, enc_bf);
  kproj_kernel<<<(B_ * S_) / 16, 256, 0, stream>>>(enc_bf, WkW, Wkb, kproj);
  hd_kernel<<<B_ / 4, 256, 0, stream>>>(hf, hr, cf, cr, hbW, hbB, cbW, cbB,
                                        h_dec, c_dec, curr);
  for (int t = 0; t < S_; ++t) {
    score_kernel<<<B_, 256, 0, stream>>>(h_dec, WqW, Wqb, kproj, vW, vb, ebuf);
    ctx_kernel<<<B_, 128, 0, stream>>>(ebuf, enc, ctx);
    cell_kernel<<<B_, 256, 0, stream>>>(ctx, dWih, dWhh, db, fcW, fcb,
                                        h_dec, c_dec, curr, out, t);
  }
}
